// MVNMPNN_40656160424013
// MI455X (gfx1250) — matrix operand order repaired and checked by reading, not compiled, not run on hardware
//
#include <hip/hip_runtime.h>

#define HS 128
#define HV 16

typedef __attribute__((ext_vector_type(16))) _Float16 v16h;
typedef __attribute__((ext_vector_type(8)))  float    v8f;

// grade of blades [1, e1, e2, e3, e12, e13, e23, e123]
constexpr int kGrade[8] = {0,1,1,1,2,2,2,3};

// Cl(3,0) geometric product tables: result blade index and sign for e_a * e_b
constexpr int GP_IDX[8][8] = {
  {0,1,2,3,4,5,6,7},
  {1,0,4,5,2,3,7,6},
  {2,4,0,6,1,7,3,5},
  {3,5,6,0,7,1,2,4},
  {4,2,1,7,0,6,5,3},
  {5,3,7,1,6,0,4,2},
  {6,7,3,2,5,4,0,1},
  {7,6,5,4,3,2,1,0}};
constexpr float GP_SGN[8][8] = {
  { 1, 1, 1, 1, 1, 1, 1, 1},
  { 1, 1, 1, 1, 1, 1, 1, 1},
  { 1,-1, 1, 1,-1,-1, 1,-1},
  { 1,-1,-1, 1, 1,-1,-1, 1},
  { 1,-1, 1, 1,-1,-1, 1,-1},
  { 1,-1,-1, 1, 1,-1,-1, 1},
  { 1, 1,-1, 1,-1, 1,-1,-1},
  { 1, 1,-1, 1,-1, 1,-1,-1}};

__device__ __forceinline__ v8f wmma16(v16h a, v16h b, v8f c) {
  return __builtin_amdgcn_wmma_f32_16x16x32_f16(
      false, a, false, b, (short)0, c, false, false);
}

// ---- A-fragment loaders (f16 A 16x32 in this code's contiguous k order; lane holds row m=lane&15,
// halves h -> K=side*16+h) ----
__device__ __forceinline__ v16h ldsA_f16(_Float16* base, int ld, int k0,
                                         int sub, int side) {
  v16h a;
#pragma unroll
  for (int h = 0; h < 16; ++h) {
    int K = k0 + side*16 + h;
    a[h] = base[sub*ld + K];
  }
  return a;
}

// state [16 rows][16 ch][8 blades] f32, K=16 channels zero-padded to 32
__device__ __forceinline__ v16h ldsA_state(float (*st)[16][8], int kb,
                                           int sub, int side) {
  v16h a;
#pragma unroll
  for (int h = 0; h < 16; ++h) {
    if (side == 0) a[h] = (_Float16)st[sub][h][kb];
    else           a[h] = (_Float16)0.f;
  }
  return a;
}

// cat [16 rows][32 ch][8 blades] f32, full K=32
__device__ __forceinline__ v16h ldsA_cat32(float (*vc)[32][8], int kb,
                                           int sub, int side) {
  v16h a;
#pragma unroll
  for (int h = 0; h < 16; ++h) {
    int c = side*16 + h;
    a[h] = (_Float16)vc[sub][c][kb];
  }
  return a;
}

// ---- B-fragment loaders (f16 B 32x16: lane n=lane&15, K=(lane>>4)*16+h) ----
// W [16 out][16 in], K padded 16->32 with zeros
__device__ __forceinline__ v16h gB_w16(const float* W, int sub, int side) {
  v16h b;
#pragma unroll
  for (int h = 0; h < 16; ++h)
    b[h] = side ? (_Float16)0.f : (_Float16)W[sub*16 + h];
  return b;
}
// W [16 out][32 in], full K=32
__device__ __forceinline__ v16h gB_w32(const float* W, int sub, int side) {
  v16h b;
#pragma unroll
  for (int h = 0; h < 16; ++h)
    b[h] = (_Float16)W[sub*32 + side*16 + h];
  return b;
}
// vlin_w [16 out][32 in][4 grades], grade g, full K=32
__device__ __forceinline__ v16h gB_vlin(const float* W, int g, int sub, int side) {
  v16h b;
#pragma unroll
  for (int h = 0; h < 16; ++h)
    b[h] = (_Float16)W[((sub*32) + (side*16 + h))*4 + g];
  return b;
}
// ev_wout [16 out][16 in][4 grades], grade g, K padded 16->32
__device__ __forceinline__ v16h gB_wout(const float* W, int g, int sub, int side) {
  v16h b;
#pragma unroll
  for (int h = 0; h < 16; ++h)
    b[h] = side ? (_Float16)0.f : (_Float16)W[(sub*16 + h)*4 + g];
  return b;
}

// Generic 16-row MLP layer: f16 LDS input [16][ldin] -> 16x(Nt*16) output.
__device__ __forceinline__ void mlp16(_Float16* in, int ldin, int Kt,
                                      const _Float16* pk, int Nt,
                                      const float* bias, bool dorelu,
                                      _Float16* out16, float* out32, int ldout,
                                      int lane, int sub, int side) {
  const v16h* pkv = (const v16h*)pk;
  for (int nt = 0; nt < Nt; ++nt) {
    v8f acc = {};
    for (int kc = 0; kc < Kt; ++kc) {
      v16h a = ldsA_f16(in, ldin, kc*32, sub, side);
      v16h b = pkv[(nt*Kt + kc)*32 + lane];
      acc = wmma16(a, b, acc);
    }
    int n = nt*16 + sub;
    float bv = bias[n];
#pragma unroll
    for (int g = 0; g < 8; ++g) {
      float val = acc[g] + bv;
      if (dorelu) val = fmaxf(val, 0.f);
      int m = side*8 + g;
      if (out16) out16[m*ldout + n] = (_Float16)val;
      if (out32) out32[m*ldout + n] = val;
    }
  }
}

// ---- Weight packing: W [Nout][Kin] f32 -> f16 B-panels in fragment order ----
__global__ void pack_b(const float* W, _Float16* dst, int Nout, int Kin, int total) {
  int t = blockIdx.x * blockDim.x + threadIdx.x;
  if (t >= total) return;
  int Kt = (Kin + 31) / 32;
  int h  = t & 15;
  int L  = (t >> 4) & 31;
  int fc = t >> 9;
  int kc = fc % Kt;
  int nt = fc / Kt;
  int n  = nt*16 + (L & 15);
  int K  = kc*32 + (L >> 4)*16 + h;
  dst[t] = (K < Kin) ? (_Float16)W[(long)n*Kin + K] : (_Float16)0.f;
}

__global__ void count_deg(const int* ej, float* deg, int E) {
  int t = blockIdx.x * blockDim.x + threadIdx.x;
  if (t < E) atomicAdd(&deg[ej[t]], 1.0f);
}

// ======================= EDGE KERNEL (1 wave = 16 edges) =======================
struct EdgeP {
  const float *s, *v, *pos;
  const int *ei, *ej;
  const float *vlin_w, *vlin_b;
  const float *ev_w0, *ev_b0, *ev_a0q, *ev_a0k;
  const float *ev_w1, *ev_b1, *ev_a1q, *ev_a1k;
  const float *ev_wout, *ev_bout;
  const float *es_b1, *es_b2, *pn_b1, *pn_b2;
  const float *mn_g, *mn_b, *mn_a;
  const _Float16 *pk_es1, *pk_es2, *pk_pn1, *pk_pn2;
  float *s_agg, *v_agg;
  int E;
};

__global__ void __launch_bounds__(32) edge_kernel(EdgeP P) {
  __shared__ float    sh_vcat[16][32][8];   // concat(v[i],v[j])
  __shared__ float    sh_state[16][16][8];  // multivector pipeline state
  __shared__ float    sh_lin[16][16][8];
  __shared__ float    sh_q[16][16][8];
  __shared__ float    sh_k[16][16][8];
  __shared__ _Float16 sh_sij[16][288];      // s_ij rows, K-padded 273->288
  __shared__ _Float16 sh_h1[16][128];
  __shared__ _Float16 sh_sm16[16][128];
  __shared__ float    sh_smes[16][128];
  __shared__ float    sh_pos[16][64];
  __shared__ float    sh_nrm[16][16];
  __shared__ float    sh_mu[16], sh_rs[16], sh_mean[16];
  __shared__ int      sh_i[16], sh_j[16];

  const int lane = threadIdx.x, sub = lane & 15, side = lane >> 4;
  const long e0 = (long)blockIdx.x * 16;

  if (lane < 16) {
    long e = e0 + lane; if (e >= P.E) e = P.E - 1;
    sh_i[lane] = P.ei[e]; sh_j[lane] = P.ej[e];
  }
  __syncthreads();

  // ---- gather v[i],v[j] and s[i],s[j],dpos into LDS ----
  {
    int m = sub;
    int node = side ? sh_j[m] : sh_i[m];
    const float* vp = P.v + (long)node * (HV*8);
#pragma unroll
    for (int c16 = 0; c16 < 16; ++c16) {
      float4 lo = ((const float4*)vp)[c16*2];
      float4 hi = ((const float4*)vp)[c16*2 + 1];
      int c = side*16 + c16;
      *(float4*)&sh_vcat[m][c][0] = lo;
      *(float4*)&sh_vcat[m][c][4] = hi;
    }
    const float* sp = P.s + (long)node * HS;
#pragma unroll
    for (int c = 0; c < HS; c += 4) {
      float4 x = *(const float4*)(sp + c);
      int base = side*128 + c;
      sh_sij[m][base+0] = (_Float16)x.x;
      sh_sij[m][base+1] = (_Float16)x.y;
      sh_sij[m][base+2] = (_Float16)x.z;
      sh_sij[m][base+3] = (_Float16)x.w;
    }
    if (side == 0) {
      sh_sij[m][256] = (_Float16)(P.pos[sh_i[m]] - P.pos[sh_j[m]]);
#pragma unroll
      for (int t = 273; t < 288; ++t) sh_sij[m][t] = (_Float16)0.f;
    }
  }
  __syncthreads();

  // ---- v_ij = mv_linear(vcat, vlin_w, subspaces) : 8 WMMAs, K=32 exact ----
  {
    float invacc[8] = {};
#pragma unroll
    for (int kb = 0; kb < 8; ++kb) {
      v16h a = ldsA_cat32(sh_vcat, kb, sub, side);
      v16h b = gB_vlin(P.vlin_w, kGrade[kb], sub, side);
      v8f d = {};
      d = wmma16(a, b, d);
      float bv = (kb == 0) ? P.vlin_b[sub] : 0.f;
#pragma unroll
      for (int g = 0; g < 8; ++g) {
        float val = d[g] + bv;
        sh_state[side*8 + g][sub][kb] = val;
        invacc[g] += val * val;
      }
    }
#pragma unroll
    for (int g = 0; g < 8; ++g)
      sh_sij[side*8 + g][257 + sub] = (_Float16)invacc[g];
  }
  __syncthreads();

  // ---- scalar edge MLP: 288->128 relu, 128->128 ----
  mlp16(&sh_sij[0][0], 288, 9, P.pk_es1, 8, P.es_b1, true,
        &sh_h1[0][0], nullptr, 128, lane, sub, side);
  __syncthreads();
  mlp16(&sh_h1[0][0], 128, 4, P.pk_es2, 8, P.es_b2, false,
        &sh_sm16[0][0], &sh_smes[0][0], 128, lane, sub, side);
  __syncthreads();
  // ---- pos net on raw s_mes: 128->128 relu, 128->64 ----
  mlp16(&sh_sm16[0][0], 128, 4, P.pk_pn1, 8, P.pn_b1, true,
        &sh_h1[0][0], nullptr, 128, lane, sub, side);
  __syncthreads();
  mlp16(&sh_h1[0][0], 128, 4, P.pk_pn2, 4, P.pn_b2, false,
        nullptr, &sh_pos[0][0], 64, lane, sub, side);
  __syncthreads();

  // ---- layer_norm(s_mes) stats ----
  if (lane < 16) {
    float mu = 0.f;
    for (int c = 0; c < 128; ++c) mu += sh_smes[lane][c];
    mu *= (1.f/128.f);
    float var = 0.f;
    for (int c = 0; c < 128; ++c) { float dd = sh_smes[lane][c] - mu; var += dd*dd; }
    var *= (1.f/128.f);
    sh_mu[lane] = mu; sh_rs[lane] = rsqrtf(var + 1e-5f);
  }
  __syncthreads();
  // ---- scatter s messages ----
  for (int m = 0; m < 16; ++m) {
    if (e0 + m >= P.E) break;
    int dst = sh_j[m];
    float mu = sh_mu[m], rs = sh_rs[m];
#pragma unroll
    for (int t = 0; t < 4; ++t) {
      int c = lane + t*32;
      float val = (sh_smes[m][c] - mu)*rs*P.mn_g[c] + P.mn_b[c];
      atomicAdd(&P.s_agg[(long)dst*HS + c], val);
    }
  }

  // ---- two mv_act(mv_linear(...)) layers ----
  const float* Wl[2] = {P.ev_w0,  P.ev_w1};
  const float* Bl[2] = {P.ev_b0,  P.ev_b1};
  const float* Wq[2] = {P.ev_a0q, P.ev_a1q};
  const float* Wk[2] = {P.ev_a0k, P.ev_a1k};
  for (int l = 0; l < 2; ++l) {
    v16h bw = gB_w16(Wl[l], sub, side);
#pragma unroll
    for (int kb = 0; kb < 8; ++kb) {
      v16h a = ldsA_state(sh_state, kb, sub, side);
      v8f d = {};
      d = wmma16(a, bw, d);
      float bv = (kb == 0) ? Bl[l][sub] : 0.f;
#pragma unroll
      for (int g = 0; g < 8; ++g) sh_lin[side*8 + g][sub][kb] = d[g] + bv;
    }
    __syncthreads();
    v16h bq = gB_w16(Wq[l], sub, side);
    v16h bk = gB_w16(Wk[l], sub, side);
    float qk[8] = {}, ks[8] = {};
#pragma unroll
    for (int kb = 0; kb < 8; ++kb) {
      v16h a = ldsA_state(sh_lin, kb, sub, side);
      v8f dq = {}; dq = wmma16(a, bq, dq);
      v8f dk = {}; dk = wmma16(a, bk, dk);
#pragma unroll
      for (int g = 0; g < 8; ++g) {
        int m = side*8 + g;
        sh_q[m][sub][kb] = dq[g];
        sh_k[m][sub][kb] = dk[g];
        qk[g] += dq[g]*dk[g];
        ks[g] += dk[g]*dk[g];
      }
    }
#pragma unroll
    for (int g = 0; g < 8; ++g) {
      int m = side*8 + g;
      float invv = qk[g] / ks[g];
      bool gate = qk[g] > 0.f;
#pragma unroll
      for (int kb = 0; kb < 8; ++kb) {
        float qv = sh_q[m][sub][kb], kv = sh_k[m][sub][kb];
        sh_state[m][sub][kb] = gate ? qv : (qv - invv*kv);
      }
    }
    __syncthreads();
  }

  // ---- v_mes = mv_linear(h, ev_wout, subspaces) * pos_msg[grade] ----
#pragma unroll
  for (int kb = 0; kb < 8; ++kb) {
    v16h a = ldsA_state(sh_state, kb, sub, side);
    v16h b = gB_wout(P.ev_wout, kGrade[kb], sub, side);
    v8f d = {};
    d = wmma16(a, b, d);
    float bv = (kb == 0) ? P.ev_bout[sub] : 0.f;
#pragma unroll
    for (int g = 0; g < 8; ++g) {
      int m = side*8 + g;
      sh_lin[m][sub][kb] = (d[g] + bv) * sh_pos[m][sub*4 + kGrade[kb]];
    }
  }
  __syncthreads();

  // ---- mv_layernorm + scatter v messages ----
#pragma unroll
  for (int t = 0; t < 8; ++t) {
    int p = lane + 32*t, m = p >> 4, o = p & 15;
    float acc = 0.f;
#pragma unroll
    for (int kb = 0; kb < 8; ++kb) { float x = sh_lin[m][o][kb]; acc += x*x; }
    sh_nrm[m][o] = sqrtf(acc);
  }
  __syncthreads();
  if (lane < 16) {
    float acc = 0.f;
    for (int o = 0; o < 16; ++o) acc += sh_nrm[lane][o];
    sh_mean[lane] = acc*(1.f/16.f) + 1e-6f;
  }
  __syncthreads();
#pragma unroll
  for (int t = 0; t < 8; ++t) {
    int p = lane + 32*t, m = p >> 4, o = p & 15;
    if (e0 + m >= P.E) continue;
    int dst = sh_j[m];
    float scale = P.mn_a[o] / sh_mean[m];
#pragma unroll
    for (int kb = 0; kb < 8; ++kb)
      atomicAdd(&P.v_agg[(long)dst*(HV*8) + o*8 + kb], sh_lin[m][o][kb]*scale);
  }
}

// ======================= NODE KERNEL (1 wave = 16 nodes) =======================
struct NodeP {
  const float *s, *v;
  const float *s_agg, *v_agg, *deg;
  const float *uv_wl, *uv_bl, *uv_wr, *uv_br, *uv_wo, *uv_bo, *uv_a;
  const float *us_b1, *us_b2, *un_g, *un_b;
  const _Float16 *pk_us1, *pk_us2;
  float *out_s, *out_v;
  int N;
};

__global__ void __launch_bounds__(32) node_kernel(NodeP P) {
  __shared__ float    sh_state[16][16][8];   // scaled v_agg
  __shared__ float    sh_vl[16][16][8];
  __shared__ float    sh_vr[16][16][8];
  __shared__ float    sh_cat[16][32][8];     // [gpv ; v_agg]
  __shared__ float    sh_out[16][16][8];
  __shared__ _Float16 sh_sa[16][128];
  __shared__ _Float16 sh_h1[16][128];
  __shared__ float    sh_su[16][128];
  __shared__ float    sh_nrm[16][16];
  __shared__ float    sh_mu[16], sh_rs[16], sh_mean[16], sh_is[16];

  const int lane = threadIdx.x, sub = lane & 15, side = lane >> 4;
  const long n0 = (long)blockIdx.x * 16;

  if (lane < 16) {
    long n = n0 + lane; if (n >= P.N) n = P.N - 1;
    float dg = P.deg[n];
    sh_is[lane] = rsqrtf(dg > 0.f ? dg : 1.f);
  }
  __syncthreads();
  {
    int m = sub; long n = n0 + m; if (n >= P.N) n = P.N - 1;
    float is = sh_is[m];
    const float* va = P.v_agg + n*(HV*8);
#pragma unroll
    for (int t = 0; t < 8; ++t) {
      int o = side*8 + t;
#pragma unroll
      for (int kb = 0; kb < 8; ++kb) sh_state[m][o][kb] = va[o*8 + kb]*is;
    }
    const float* sa = P.s_agg + n*HS;
#pragma unroll
    for (int c = 0; c < 64; ++c)
      sh_sa[m][side*64 + c] = (_Float16)(sa[side*64 + c]*is);
  }
  __syncthreads();

  // ---- vl, vr = mv_linear(v_agg, uv_wl/uv_wr) ----
  {
    v16h bl = gB_w16(P.uv_wl, sub, side);
    v16h br = gB_w16(P.uv_wr, sub, side);
#pragma unroll
    for (int kb = 0; kb < 8; ++kb) {
      v16h a = ldsA_state(sh_state, kb, sub, side);
      v8f dl = {}; dl = wmma16(a, bl, dl);
      v8f dr = {}; dr = wmma16(a, br, dr);
      float bvl = (kb == 0) ? P.uv_bl[sub] : 0.f;
      float bvr = (kb == 0) ? P.uv_br[sub] : 0.f;
#pragma unroll
      for (int g = 0; g < 8; ++g) {
        sh_vl[side*8 + g][sub][kb] = dl[g] + bvl;
        sh_vr[side*8 + g][sub][kb] = dr[g] + bvr;
      }
    }
  }
  __syncthreads();

  // ---- geometric product (Cayley) + concat with v_agg ----
#pragma unroll
  for (int t = 0; t < 8; ++t) {
    int p = lane + 32*t, m = p >> 4, o = p & 15;
    float av[8], bv[8], out[8] = {};
#pragma unroll
    for (int kb = 0; kb < 8; ++kb) { av[kb] = sh_vl[m][o][kb]; bv[kb] = sh_vr[m][o][kb]; }
#pragma unroll
    for (int x = 0; x < 8; ++x)
#pragma unroll
      for (int y = 0; y < 8; ++y)
        out[GP_IDX[x][y]] += GP_SGN[x][y]*av[x]*bv[y];
#pragma unroll
    for (int kb = 0; kb < 8; ++kb) {
      sh_cat[m][o][kb]      = out[kb];
      sh_cat[m][16+o][kb]   = sh_state[m][o][kb];
    }
  }
  __syncthreads();

  // ---- v_out = mv_linear([gpv; v_agg], uv_wo) : K=32 exact ----
#pragma unroll
  for (int kb = 0; kb < 8; ++kb) {
    v16h a = ldsA_cat32(sh_cat, kb, sub, side);
    v16h b = gB_w32(P.uv_wo, sub, side);
    v8f d = {};
    d = wmma16(a, b, d);
    float bvv = (kb == 0) ? P.uv_bo[sub] : 0.f;
#pragma unroll
    for (int g = 0; g < 8; ++g) sh_out[side*8 + g][sub][kb] = d[g] + bvv;
  }
  __syncthreads();

  // ---- mv_layernorm(uv_a) + residual -> out_v ----
#pragma unroll
  for (int t = 0; t < 8; ++t) {
    int p = lane + 32*t, m = p >> 4, o = p & 15;
    float acc = 0.f;
#pragma unroll
    for (int kb = 0; kb < 8; ++kb) { float x = sh_out[m][o][kb]; acc += x*x; }
    sh_nrm[m][o] = sqrtf(acc);
  }
  __syncthreads();
  if (lane < 16) {
    float acc = 0.f;
    for (int o = 0; o < 16; ++o) acc += sh_nrm[lane][o];
    sh_mean[lane] = acc*(1.f/16.f) + 1e-6f;
  }
  __syncthreads();
#pragma unroll
  for (int t = 0; t < 8; ++t) {
    int p = lane + 32*t, m = p >> 4, o = p & 15;
    long n = n0 + m;
    if (n >= P.N) continue;
    float scale = P.uv_a[o] / sh_mean[m];
#pragma unroll
    for (int kb = 0; kb < 8; ++kb) {
      long idx = n*(HV*8) + o*8 + kb;
      P.out_v[idx] = sh_out[m][o][kb]*scale + P.v[idx];
    }
  }

  // ---- scalar node MLP: 128->128 relu, 128->128, layernorm, residual ----
  mlp16(&sh_sa[0][0], 128, 4, P.pk_us1, 8, P.us_b1, true,
        &sh_h1[0][0], nullptr, 128, lane, sub, side);
  __syncthreads();
  mlp16(&sh_h1[0][0], 128, 4, P.pk_us2, 8, P.us_b2, false,
        nullptr, &sh_su[0][0], 128, lane, sub, side);
  __syncthreads();
  if (lane < 16) {
    float mu = 0.f;
    for (int c = 0; c < 128; ++c) mu += sh_su[lane][c];
    mu *= (1.f/128.f);
    float var = 0.f;
    for (int c = 0; c < 128; ++c) { float dd = sh_su[lane][c] - mu; var += dd*dd; }
    var *= (1.f/128.f);
    sh_mu[lane] = mu; sh_rs[lane] = rsqrtf(var + 1e-5f);
  }
  __syncthreads();
  for (int m = 0; m < 16; ++m) {
    long n = n0 + m;
    if (n >= P.N) break;
    float mu = sh_mu[m], rs = sh_rs[m];
#pragma unroll
    for (int t = 0; t < 4; ++t) {
      int c = lane + t*32;
      long idx = n*HS + c;
      P.out_s[idx] = (sh_su[m][c] - mu)*rs*P.un_g[c] + P.un_b[c] + P.s[idx];
    }
  }
}

// ======================= HOST LAUNCH =======================
extern "C" void kernel_launch(void* const* d_in, const int* in_sizes, int n_in,
                              void* d_out, int out_size, void* d_ws, size_t ws_size,
                              hipStream_t stream) {
  const float* s   = (const float*)d_in[0];
  const float* v   = (const float*)d_in[1];
  const float* pos = (const float*)d_in[2];
  const int* eidx  = (const int*)d_in[3];
  const int E = in_sizes[3] / 2;
  const int N = in_sizes[2];
  const int* ei = eidx;
  const int* ej = eidx + E;

  // params in setup_inputs() dict order, after the 4 tensors
  const float* prm[36];
  for (int i = 0; i < 36; ++i) prm[i] = (const float*)d_in[4 + i];
  const float *vlin_w = prm[0],  *vlin_b = prm[1];
  const float *ev_w0  = prm[2],  *ev_b0  = prm[3],  *ev_a0q = prm[4],  *ev_a0k = prm[5];
  const float *ev_w1  = prm[6],  *ev_b1  = prm[7],  *ev_a1q = prm[8],  *ev_a1k = prm[9];
  const float *ev_wout= prm[10], *ev_bout= prm[11];
  const float *es_w1  = prm[12], *es_b1  = prm[13], *es_w2 = prm[14], *es_b2 = prm[15];
  const float *pn_w1  = prm[16], *pn_b1  = prm[17], *pn_w2 = prm[18], *pn_b2 = prm[19];
  const float *mn_g   = prm[20], *mn_b   = prm[21], *mn_a  = prm[22];
  const float *uv_wl  = prm[23], *uv_bl  = prm[24], *uv_wr = prm[25], *uv_br = prm[26];
  const float *uv_wo  = prm[27], *uv_bo  = prm[28], *uv_a  = prm[29];
  const float *us_w1  = prm[30], *us_b1  = prm[31], *us_w2 = prm[32], *us_b2 = prm[33];
  const float *un_g   = prm[34], *un_b   = prm[35];

  // workspace layout
  char* ws = (char*)d_ws;
  size_t off = 0;
  float* s_agg = (float*)(ws + off); off += (size_t)N*HS*sizeof(float);
  float* v_agg = (float*)(ws + off); off += (size_t)N*HV*8*sizeof(float);
  float* deg   = (float*)(ws + off); off += (size_t)N*sizeof(float);
  const size_t zero_bytes = off;
  off = (off + 255) & ~(size_t)255;
  // packed f16 weight panels
  const int sz_es1 = 8*9*512, sz_128 = 8*4*512, sz_pn2 = 4*4*512;
  _Float16* pk_es1 = (_Float16*)(ws + off); off += (size_t)sz_es1*2;
  _Float16* pk_es2 = (_Float16*)(ws + off); off += (size_t)sz_128*2;
  _Float16* pk_pn1 = (_Float16*)(ws + off); off += (size_t)sz_128*2;
  _Float16* pk_pn2 = (_Float16*)(ws + off); off += (size_t)sz_pn2*2;
  _Float16* pk_us1 = (_Float16*)(ws + off); off += (size_t)sz_128*2;
  _Float16* pk_us2 = (_Float16*)(ws + off); off += (size_t)sz_128*2;

  hipMemsetAsync(ws, 0, zero_bytes, stream);

  pack_b<<<(sz_es1 + 255)/256, 256, 0, stream>>>(es_w1, pk_es1, 128, 273, sz_es1);
  pack_b<<<(sz_128 + 255)/256, 256, 0, stream>>>(es_w2, pk_es2, 128, 128, sz_128);
  pack_b<<<(sz_128 + 255)/256, 256, 0, stream>>>(pn_w1, pk_pn1, 128, 128, sz_128);
  pack_b<<<(sz_pn2 + 255)/256, 256, 0, stream>>>(pn_w2, pk_pn2, 64, 128, sz_pn2);
  pack_b<<<(sz_128 + 255)/256, 256, 0, stream>>>(us_w1, pk_us1, 128, 128, sz_128);
  pack_b<<<(sz_128 + 255)/256, 256, 0, stream>>>(us_w2, pk_us2, 128, 128, sz_128);

  count_deg<<<(E + 255)/256, 256, 0, stream>>>(ej, deg, E);

  EdgeP ep;
  ep.s = s; ep.v = v; ep.pos = pos; ep.ei = ei; ep.ej = ej;
  ep.vlin_w = vlin_w; ep.vlin_b = vlin_b;
  ep.ev_w0 = ev_w0; ep.ev_b0 = ev_b0; ep.ev_a0q = ev_a0q; ep.ev_a0k = ev_a0k;
  ep.ev_w1 = ev_w1; ep.ev_b1 = ev_b1; ep.ev_a1q = ev_a1q; ep.ev_a1k = ev_a1k;
  ep.ev_wout = ev_wout; ep.ev_bout = ev_bout;
  ep.es_b1 = es_b1; ep.es_b2 = es_b2; ep.pn_b1 = pn_b1; ep.pn_b2 = pn_b2;
  ep.mn_g = mn_g; ep.mn_b = mn_b; ep.mn_a = mn_a;
  ep.pk_es1 = pk_es1; ep.pk_es2 = pk_es2; ep.pk_pn1 = pk_pn1; ep.pk_pn2 = pk_pn2;
  ep.s_agg = s_agg; ep.v_agg = v_agg; ep.E = E;
  edge_kernel<<<(E + 15)/16, 32, 0, stream>>>(ep);

  NodeP np;
  np.s = s; np.v = v; np.s_agg = s_agg; np.v_agg = v_agg; np.deg = deg;
  np.uv_wl = uv_wl; np.uv_bl = uv_bl; np.uv_wr = uv_wr; np.uv_br = uv_br;
  np.uv_wo = uv_wo; np.uv_bo = uv_bo; np.uv_a = uv_a;
  np.us_b1 = us_b1; np.us_b2 = us_b2; np.un_g = un_g; np.un_b = un_b;
  np.pk_us1 = pk_us1; np.pk_us2 = pk_us2;
  np.out_s = (float*)d_out;
  np.out_v = (float*)d_out + (size_t)N*HS;
  np.N = N;
  node_kernel<<<(N + 15)/16, 32, 0, stream>>>(np);
}
